// EnhancedSpatialAttention_9998683865816
// MI455X (gfx1250) — matrix operand order repaired and checked by reading, not compiled, not run on hardware
//
#include <hip/hip_runtime.h>

// CDNA5 / gfx1250 WMMA attention block, TDM-staged weights:
//   prep kernels: convert+transpose weights to f16 LDS-image slices in d_ws (once)
//   main kernel : one workgroup (256 thr = 8 wave32) per sequence; weight slices
//                 DMA'd into LDS via TENSOR_LOAD_TO_LDS (double-buffered),
//                 all GEMMs via v_wmma_f32_16x16x32_f16.

typedef __attribute__((ext_vector_type(16))) _Float16 v16h;
typedef __attribute__((ext_vector_type(8)))  _Float16 v8h;
typedef __attribute__((ext_vector_type(4)))  _Float16 v4h;
typedef __attribute__((ext_vector_type(8)))  float    v8f;
typedef __attribute__((ext_vector_type(4)))  unsigned int u32x4;
typedef __attribute__((ext_vector_type(8)))  int      i32x8;
typedef __attribute__((ext_vector_type(4)))  int      i32x4;

#define TC 64     // sequence length (channels C)
#define TD 512    // model dim D
#define NH 8      // heads
#define HDIM 64   // head dim
#define SLICE_BYTES 65536          // one wT slice: [512][64] f16
#define SLICE_ELEMS 32768
#define NSLICE_QKV 24              // 8 heads x {q,k,v}
#define NSLICE_OUT 8
#define WS_W_ELEMS (SLICE_ELEMS * (NSLICE_QKV + NSLICE_OUT))   // 1,048,576 f16
#define WS_BIAS_OFF ((size_t)WS_W_ELEMS * 2)                   // bytes

// ---------------- LDS map (dynamic, 313,344 B total) ----------------
#define L_XS   0u        // [64][512] f16  x (A operand + residual)
#define L_WT0  65536u    // [512][64] f16  weight slice buffer 0
#define L_WT1  131072u   // [512][64] f16  weight slice buffer 1 / outpre (phase D)
#define L_CTX  196608u   // [64][512] f16  attention context
#define L_QS   262144u   // [64][64] f16
#define L_KT   270336u   // [64(hd)][64(key)] f16
#define L_VS   278528u   // [64][64] f16
#define L_PS   286720u   // [64][64] f16 probs
#define L_SS   294912u   // [64][64] f32 scores
#define L_RED  311296u   // [64][8] f32 LN partials
#define SMEM_BYTES 313344u

__device__ __forceinline__ v8f wmma16(v16h a, v16h b, v8f c) {
  return __builtin_amdgcn_wmma_f32_16x16x32_f16(
      false, a, false, b, (short)0, c, false, false);
}

// A fragment: 16x32 f16 at (m0,k0) from row-major [M][ld] f16 in LDS.
__device__ __forceinline__ v16h ld_a(const _Float16* S, int ld, int m0, int k0, int lane) {
  int m  = m0 + (lane & 15);
  int kb = k0 + ((lane & 16) >> 1);   // +8 for lanes 16..31
  const _Float16* p = S + m * ld + kb;
  v8h lo = *(const v8h*)(p);
  v8h hi = *(const v8h*)(p + 16);
  v16h r;
#pragma unroll
  for (int i = 0; i < 8; ++i) { r[i] = lo[i]; r[i + 8] = hi[i]; }
  return r;
}

// B fragment: 32x16 f16 at (k0,n0) from K-major [K][ld] f16 in LDS.
__device__ __forceinline__ v16h ld_b(const _Float16* S, int ld, int k0, int n0, int lane) {
  int n  = n0 + (lane & 15);
  int kb = k0 + ((lane & 16) >> 1);
  v16h r;
#pragma unroll
  for (int i = 0; i < 8; ++i) { r[i] = S[(kb + i) * ld + n]; r[i + 8] = S[(kb + 16 + i) * ld + n]; }
  return r;
}

__device__ __forceinline__ void st_d_f16(_Float16* D, int ld, int m0, int n0, int lane, v8f acc) {
  int n  = n0 + (lane & 15);
  int mb = m0 + ((lane & 16) >> 1);
#pragma unroll
  for (int r = 0; r < 8; ++r) D[(mb + r) * ld + n] = (_Float16)acc[r];
}
__device__ __forceinline__ void st_d_f16_T(_Float16* D, int ld, int m0, int n0, int lane, v8f acc) {
  int n  = n0 + (lane & 15);
  int mb = m0 + ((lane & 16) >> 1);
#pragma unroll
  for (int r = 0; r < 8; ++r) D[n * ld + (mb + r)] = (_Float16)acc[r];
}
__device__ __forceinline__ void st_d_f32(float* D, int ld, int m0, int n0, int lane, v8f acc) {
  int n  = n0 + (lane & 15);
  int mb = m0 + ((lane & 16) >> 1);
#pragma unroll
  for (int r = 0; r < 8; ++r) D[(mb + r) * ld + n] = acc[r];
}

// 64x64xK GEMM: 8 waves cover the 4x4 tile grid, 2 tiles per wave.
__device__ __forceinline__ void gemm64(const _Float16* __restrict__ A, int lda,
                                       const _Float16* __restrict__ B, int ldb,
                                       int K, int wave, int lane,
                                       v8f& acc0, v8f& acc1, int& mt, int& nt0) {
  mt  = wave >> 1;
  nt0 = (wave & 1) << 1;
  v8f z = {0.f, 0.f, 0.f, 0.f, 0.f, 0.f, 0.f, 0.f};
  acc0 = z; acc1 = z;
  for (int k0 = 0; k0 < K; k0 += 32) {
    v16h a  = ld_a(A, lda, mt * 16, k0, lane);
    v16h b0 = ld_b(B, ldb, k0, nt0 * 16,      lane);
    v16h b1 = ld_b(B, ldb, k0, nt0 * 16 + 16, lane);
    acc0 = wmma16(a, b0, acc0);
    acc1 = wmma16(a, b1, acc1);
  }
}

// ---- TDM: DMA one 64KB f16 slice (global, final layout) -> LDS -----------
// D# per CDNA5 ISA: 1D tile, 8192 x 8-byte quanta.
__device__ __forceinline__ void tdm_issue(const void* gsrc, unsigned lds_off) {
  unsigned long long ga = (unsigned long long)(uintptr_t)gsrc;
  u32x4 g0;
  g0[0] = 1u;                                   // count=1, user mode
  g0[1] = lds_off;                              // lds_addr (bytes)
  g0[2] = (unsigned)ga;                         // global_addr[31:0]
  g0[3] = (unsigned)((ga >> 32) & 0x01FFFFFFull) | (2u << 30); // addr[56:32], type=2
  i32x8 g1 = {0, 0, 0, 0, 0, 0, 0, 0};
  g1[0] = (3 << 16);                 // workgroup_mask=0, data_size=3 (8B)
  g1[1] = (int)(8192u << 16);        // tensor_dim0[15:0]=8192 in [31:16]
  g1[2] = (1 << 16);                 // tensor_dim0[31:16]=0, tensor_dim1=1
  g1[3] = (int)(8192u << 16);        // tile_dim0=8192 in [31:16]
  g1[4] = 1;                         // tile_dim1=1, tile_dim2=0
  g1[5] = 8192;                      // tensor_dim0_stride lo32
  i32x4 g2 = {0, 0, 0, 0};
  i32x4 g3 = {0, 0, 0, 0};
#if __clang_major__ >= 23
  i32x8 g4 = {0, 0, 0, 0, 0, 0, 0, 0};
  __builtin_amdgcn_tensor_load_to_lds(g0, g1, g2, g3, g4, 0);
#else
  __builtin_amdgcn_tensor_load_to_lds(g0, g1, g2, g3, 0);
#endif
}

// ---------------- prep kernels (run once per launch) ----------------------
// ws f16 image: slices s=0..23: h=s/3, t=s%3 -> in_proj rows t*512+h*64+j,
// stored transposed wT[d*64+j]; slices 24..31: out_proj rows cc*64+j.
__global__ __launch_bounds__(256)
void esa_prep_weights(const float* __restrict__ ipw, const float* __restrict__ opw,
                      _Float16* __restrict__ wsW) {
  for (size_t e = (size_t)blockIdx.x * 256 + threadIdx.x; e < (size_t)WS_W_ELEMS;
       e += (size_t)gridDim.x * 256) {
    int s  = (int)(e >> 15);
    int o  = (int)(e & 32767);
    int d  = o >> 6;
    int j  = o & 63;
    int row;
    if (s < NSLICE_QKV) {
      int h = s / 3, t = s - 3 * h;
      row = t * TD + h * HDIM + j;
      wsW[e] = (_Float16)ipw[(size_t)row * TD + d];
    } else {
      int cc = s - NSLICE_QKV;
      row = cc * 64 + j;
      wsW[e] = (_Float16)opw[(size_t)row * TD + d];
    }
  }
}

__global__ __launch_bounds__(256)
void esa_prep_bias(const float* __restrict__ rpb, float* __restrict__ wsB) {
  int i = blockIdx.x * 256 + threadIdx.x;
  if (i < TC * TC) {
    int q = i >> 6, kk = i & 63;
    float s = 0.f;
#pragma unroll
    for (int h = 0; h < NH; ++h) s += rpb[(h * 128 + q) * 128 + kk];
    wsB[i] = s * (1.f / NH);
  }
}

// ---------------- main kernel ---------------------------------------------
__global__ __launch_bounds__(256)
void esa_attn_kernel(const float* __restrict__ x,
                     const _Float16* __restrict__ wsW,  // 32 staged slices
                     const float* __restrict__ wsB,     // [64][64] bias tile
                     const float* __restrict__ ipb,     // [1536]
                     const float* __restrict__ opb,     // [512]
                     const float* __restrict__ lng,     // [512]
                     const float* __restrict__ lnb,     // [512]
                     float* __restrict__ out) {
  extern __shared__ char smem[];
  _Float16* xs  = (_Float16*)(smem + L_XS);
  _Float16* wt0 = (_Float16*)(smem + L_WT0);
  _Float16* wt1 = (_Float16*)(smem + L_WT1);
  _Float16* ctx = (_Float16*)(smem + L_CTX);
  _Float16* qs  = (_Float16*)(smem + L_QS);
  _Float16* kT  = (_Float16*)(smem + L_KT);
  _Float16* vs  = (_Float16*)(smem + L_VS);
  _Float16* ps  = (_Float16*)(smem + L_PS);
  float*    ss  = (float*)(smem + L_SS);
  _Float16* outpre = (_Float16*)(smem + L_WT1);   // phase-D reuse of wT1
  float*    red = (float*)(smem + L_RED);

  const int tid  = threadIdx.x;
  const int wave = tid >> 5;
  const int lane = tid & 31;
  const int n    = blockIdx.x;
  const char* wsbytes = (const char*)wsW;

  // Kick off DMA of the first two weight slices, overlap with x staging.
  if (wave == 0) {
    tdm_issue(wsbytes + 0 * (size_t)SLICE_BYTES, L_WT0);
    tdm_issue(wsbytes + 1 * (size_t)SLICE_BYTES, L_WT1);
  }

  // Stage x as f16 (vectorized b128 reads / b64 LDS writes).
  {
    const float4* xv = (const float4*)(x + (size_t)n * TC * TD);
    for (int i = tid; i < TC * TD / 4; i += 256) {
      float4 v = xv[i];
      v4h h4 = { (_Float16)v.x, (_Float16)v.y, (_Float16)v.z, (_Float16)v.w };
      *(v4h*)(xs + i * 4) = h4;
    }
  }
  __syncthreads();

  v8f a0, a1; int mt, nt0;

  for (int h = 0; h < NH; ++h) {
#pragma unroll
    for (int t = 0; t < 3; ++t) {           // 0:q  1:k  2:v
      const int i = h * 3 + t;
      if (wave == 0) {
        if (i < NSLICE_QKV - 1) __builtin_amdgcn_s_wait_tensorcnt(1);
        else                    __builtin_amdgcn_s_wait_tensorcnt(0);
      }
      __syncthreads();                       // slice i resident in wt[i&1]
      const _Float16* wcur = (i & 1) ? wt1 : wt0;
      gemm64(xs, TD, wcur, HDIM, TD, wave, lane, a0, a1, mt, nt0);
      {
        int jb = t * TD + h * HDIM + nt0 * 16 + (lane & 15);
        float b0 = ipb[jb], b1 = ipb[jb + 16];
#pragma unroll
        for (int r = 0; r < 8; ++r) { a0[r] += b0; a1[r] += b1; }
        if (t == 0) {
          st_d_f16(qs, HDIM, mt * 16, nt0 * 16,      lane, a0);
          st_d_f16(qs, HDIM, mt * 16, nt0 * 16 + 16, lane, a1);
        } else if (t == 1) {                 // store transposed kT[hd][key]
          st_d_f16_T(kT, TC, mt * 16, nt0 * 16,      lane, a0);
          st_d_f16_T(kT, TC, mt * 16, nt0 * 16 + 16, lane, a1);
        } else {
          st_d_f16(vs, HDIM, mt * 16, nt0 * 16,      lane, a0);
          st_d_f16(vs, HDIM, mt * 16, nt0 * 16 + 16, lane, a1);
        }
      }
      __syncthreads();                       // gemm i done -> wt[(i+2)&1] free
      if (wave == 0 && i + 2 < NSLICE_QKV)
        tdm_issue(wsbytes + (size_t)(i + 2) * SLICE_BYTES, (i & 1) ? L_WT1 : L_WT0);
    }

    // scores = (q @ k^T) / sqrt(hd)
    gemm64(qs, HDIM, kT, TC, HDIM, wave, lane, a0, a1, mt, nt0);
#pragma unroll
    for (int r = 0; r < 8; ++r) { a0[r] *= 0.125f; a1[r] *= 0.125f; }
    st_d_f32(ss, TC, mt * 16, nt0 * 16,      lane, a0);
    st_d_f32(ss, TC, mt * 16, nt0 * 16 + 16, lane, a1);
    __syncthreads();

    // softmax(scores + bias): one thread per query row, bias read from L2
    if (tid < TC) {
      float* row = ss + tid * TC;
      const float* brow = wsB + tid * TC;
      float mx = -1e30f;
      for (int kk = 0; kk < TC; ++kk) {
        float v = row[kk] + brow[kk];
        row[kk] = v;
        mx = v > mx ? v : mx;
      }
      float sum = 0.f;
      for (int kk = 0; kk < TC; ++kk) {
        float e = __expf(row[kk] - mx);
        sum += e;
        row[kk] = e;
      }
      float inv = 1.f / sum;
      _Float16* prow = ps + tid * TC;
      for (int kk = 0; kk < TC; ++kk) prow[kk] = (_Float16)(row[kk] * inv);
    }
    __syncthreads();

    // ctx_h = P @ V
    gemm64(ps, TC, vs, HDIM, TC, wave, lane, a0, a1, mt, nt0);
    st_d_f16(ctx, TD, mt * 16, h * HDIM + nt0 * 16,      lane, a0);
    st_d_f16(ctx, TD, mt * 16, h * HDIM + nt0 * 16 + 16, lane, a1);
    __syncthreads();
  }

  // Phase D: out = ctx @ Wout^T + b_out + x; 8 column chunks, wT0 single-buf,
  // wT1 repurposed as f16 pre-LN buffer.
  for (int cc = 0; cc < NSLICE_OUT; ++cc) {
    if (wave == 0) {
      tdm_issue(wsbytes + (size_t)(NSLICE_QKV + cc) * SLICE_BYTES, L_WT0);
      __builtin_amdgcn_s_wait_tensorcnt(0);
    }
    __syncthreads();
    gemm64(ctx, TD, wt0, HDIM, TD, wave, lane, a0, a1, mt, nt0);
    {
      int ng0 = cc * 64 + nt0 * 16 + (lane & 15);
      int ng1 = ng0 + 16;
      int mb  = mt * 16 + ((lane & 16) >> 1);
      float ob0 = opb[ng0], ob1 = opb[ng1];
#pragma unroll
      for (int r = 0; r < 8; ++r) {
        int m = mb + r;
        outpre[m * TD + ng0] = (_Float16)(a0[r] + ob0 + (float)xs[m * TD + ng0]);
        outpre[m * TD + ng1] = (_Float16)(a1[r] + ob1 + (float)xs[m * TD + ng1]);
      }
    }
    __syncthreads();                         // gemm cc done -> wT0 free
  }

  // LayerNorm over D per row: 4 threads/row partials -> stats -> write
  {
    int r = tid & 63, p = tid >> 6;
    const _Float16* row = outpre + r * TD + p * 128;
    float sum = 0.f, sq = 0.f;
    for (int i = 0; i < 128; ++i) { float v = (float)row[i]; sum += v; sq += v * v; }
    red[r * 8 + p]     = sum;
    red[r * 8 + 4 + p] = sq;
  }
  __syncthreads();
  if (tid < TC) {
    int r = tid;
    float sum = red[r * 8] + red[r * 8 + 1] + red[r * 8 + 2] + red[r * 8 + 3];
    float sq  = red[r * 8 + 4] + red[r * 8 + 5] + red[r * 8 + 6] + red[r * 8 + 7];
    float mu  = sum * (1.f / TD);
    float var = sq * (1.f / TD) - mu * mu;
    red[r * 8]     = mu;
    red[r * 8 + 1] = rsqrtf(var + 1e-5f);
  }
  __syncthreads();
  {
    float4* on = (float4*)(out + (size_t)n * TC * TD);
    for (int i = tid; i < TC * TD / 4; i += 256) {
      int c  = (i * 4) >> 9;
      int d0 = (i * 4) & (TD - 1);
      float mu = red[c * 8], rs = red[c * 8 + 1];
      float4 o;
      o.x = ((float)outpre[i * 4 + 0] - mu) * rs * lng[d0 + 0] + lnb[d0 + 0];
      o.y = ((float)outpre[i * 4 + 1] - mu) * rs * lng[d0 + 1] + lnb[d0 + 1];
      o.z = ((float)outpre[i * 4 + 2] - mu) * rs * lng[d0 + 2] + lnb[d0 + 2];
      o.w = ((float)outpre[i * 4 + 3] - mu) * rs * lng[d0 + 3] + lnb[d0 + 3];
      on[i] = o;
    }
  }
}

extern "C" void kernel_launch(void* const* d_in, const int* in_sizes, int n_in,
                              void* d_out, int out_size, void* d_ws, size_t ws_size,
                              hipStream_t stream) {
  const float* x   = (const float*)d_in[0];
  const float* ipw = (const float*)d_in[1];
  const float* ipb = (const float*)d_in[2];
  const float* opw = (const float*)d_in[3];
  const float* opb = (const float*)d_in[4];
  const float* lng = (const float*)d_in[5];
  const float* lnb = (const float*)d_in[6];
  const float* rpb = (const float*)d_in[7];
  float* out = (float*)d_out;

  _Float16* wsW = (_Float16*)d_ws;                       // 2 MB f16 weight image
  float*    wsB = (float*)((char*)d_ws + WS_BIAS_OFF);   // 16 KB bias tile

  int nseq = in_sizes[0] / (TC * TD);                    // B*T = 2048

  esa_prep_weights<<<1024, 256, 0, stream>>>(ipw, opw, wsW);
  esa_prep_bias<<<16, 256, 0, stream>>>(rpb, wsB);

  (void)hipFuncSetAttribute((const void*)esa_attn_kernel,
                            hipFuncAttributeMaxDynamicSharedMemorySize,
                            (int)SMEM_BYTES);
  esa_attn_kernel<<<nseq, 256, SMEM_BYTES, stream>>>(x, wsW, wsB, ipb, opb,
                                                     lng, lnb, out);
}
